// TFEncoderLayer_18468359373514
// MI455X (gfx1250) — matrix operand order repaired and checked by reading, not compiled, not run on hardware
//
#include <hip/hip_runtime.h>
#include <hip/hip_bf16.h>

typedef unsigned short u16;
typedef unsigned int u32;
typedef __attribute__((ext_vector_type(16))) __bf16 v16bf;
typedef __attribute__((ext_vector_type(8)))  float  v8f;

// ---------- bf16 split helpers (native RNE converts -> v_cvt bf16 ops) ----------
__device__ __forceinline__ u16 f2bf(float f) {
  union { __bf16 b; u16 u; } c; c.b = (__bf16)f; return c.u;
}
__device__ __forceinline__ float bf2f(u16 h) {
  union { u16 u; __bf16 b; } c; c.u = h; return (float)c.b;
}

// ---------- WMMA wrapper (D = A*B + C, bf16 in, f32 acc) ----------
__device__ __forceinline__ v8f wmma_bf16(v16bf a, v16bf b, v8f c) {
  return __builtin_amdgcn_wmma_f32_16x16x32_bf16(false, a, false, b, (short)0, c, false, false);
}

// A fragment (16x32 slice): lane supplies row M = lane&15.
// element e -> k = kbase + e + 8*((e>=8) + (lane>=16))  => two 16B chunks.
__device__ __forceinline__ v16bf load_frag_a(const u16* rowptr, int hl, int kbase) {
  union { v16bf v; uint4 q[2]; } u;
  const u16* p = rowptr + kbase + 8 * hl;
  u.q[0] = *(const uint4*)(p);
  u.q[1] = *(const uint4*)(p + 16);
  return u.v;
}
// B fragment (32x16 slice), LDS stored as [n][k]: lane supplies col N = lane&15,
// element e -> k = kbase + e + 8*((e>=8) + (lane>=16))  => two 16B chunks.
__device__ __forceinline__ v16bf load_frag_b(const u16* rowptr, int hl, int kbase) {
  union { v16bf v; uint4 q[2]; } u;
  const u16* p = rowptr + kbase + 8 * hl;
  u.q[0] = *(const uint4*)(p);
  u.q[1] = *(const uint4*)(p + 16);
  return u.v;
}

// ==========================================================================
// GEMM: C[M,N] = epilogue(A[M,K] * B[K,N] + bias), bf16x3 split, f32 accum.
// EPI: 0 = +bias, 1 = (acc+bias)*scale, 2 = gelu(acc+bias), 3 = acc+bias+resid
// Tiles: 128x128x32, 256 threads = 8 waves, wave -> 32x64 (2x4 WMMA tiles).
// Software-pipelined: next K-tile global loads issue before this tile's WMMAs.
// ==========================================================================
#define BM 128
#define BN 128
#define BK 32
#define LDSA 40   // ushorts per LDS row: 32 data + 8 pad (80B, 16B aligned)

template <int EPI>
__global__ __launch_bounds__(256) void gemm_bf16x3(
    const float* __restrict__ A, const float* __restrict__ B,
    const float* __restrict__ bias, const float* __restrict__ resid,
    float* __restrict__ C, int M, int N, int K, float scale)
{
  __shared__ u16 As_h[BM * LDSA];
  __shared__ u16 As_l[BM * LDSA];
  __shared__ u16 Bs_h[BN * LDSA];
  __shared__ u16 Bs_l[BN * LDSA];

  const int tid  = threadIdx.x;
  const int lane = tid & 31;
  const int wid  = tid >> 5;
  const int wm   = wid & 3;        // 4 waves along M (32 rows each)
  const int wn   = wid >> 2;       // 2 waves along N (64 cols each)
  const int hl   = (lane >> 4) & 1;
  const int ln   = lane & 15;
  const int bm0  = blockIdx.y * BM;
  const int bn0  = blockIdx.x * BN;

  // per-thread staging indices (A: 8 float4/row; B: 32 float4/k-row)
  int fa_r[4], fa_c[4], fb_r[4], fb_c[4];
#pragma unroll
  for (int i = 0; i < 4; ++i) {
    int f = tid + i * 256;
    fa_r[i] = f >> 3;  fa_c[i] = (f & 7) << 2;
    fb_r[i] = f >> 5;  fb_c[i] = (f & 31) << 2;
  }

  v8f acc[2][4];
#pragma unroll
  for (int s = 0; s < 2; ++s)
#pragma unroll
    for (int t = 0; t < 4; ++t) acc[s][t] = (v8f){};

  // ---- prologue: load first K-tile into registers ----
  float4 ra[4], rb[4];
#pragma unroll
  for (int i = 0; i < 4; ++i) {
    ra[i] = *(const float4*)(A + (size_t)(bm0 + fa_r[i]) * K + fa_c[i]);
    rb[i] = *(const float4*)(B + (size_t)fb_r[i] * N + bn0 + fb_c[i]);
  }

  for (int kt = 0; kt < K; kt += BK) {
    __syncthreads();
    // ---- stage registers -> LDS (hi/lo split) ----
#pragma unroll
    for (int i = 0; i < 4; ++i) {
      float a0[4] = {ra[i].x, ra[i].y, ra[i].z, ra[i].w};
      float b0[4] = {rb[i].x, rb[i].y, rb[i].z, rb[i].w};
#pragma unroll
      for (int j = 0; j < 4; ++j) {
        u16 h = f2bf(a0[j]);
        As_h[fa_r[i] * LDSA + fa_c[i] + j] = h;
        As_l[fa_r[i] * LDSA + fa_c[i] + j] = f2bf(a0[j] - bf2f(h));
        u16 g = f2bf(b0[j]);
        Bs_h[(fb_c[i] + j) * LDSA + fb_r[i]] = g;          // transposed [n][k]
        Bs_l[(fb_c[i] + j) * LDSA + fb_r[i]] = f2bf(b0[j] - bf2f(g));
      }
    }
    __syncthreads();

    // ---- issue next tile's global loads before compute (latency hiding) ----
    if (kt + BK < K) {
#pragma unroll
      for (int i = 0; i < 4; ++i) {
        ra[i] = *(const float4*)(A + (size_t)(bm0 + fa_r[i]) * K + (kt + BK) + fa_c[i]);
        rb[i] = *(const float4*)(B + (size_t)(kt + BK + fb_r[i]) * N + bn0 + fb_c[i]);
      }
    }

    // ---- fragments + 24 WMMAs ----
    v16bf ah[2], al[2];
#pragma unroll
    for (int s = 0; s < 2; ++s) {
      int row = wm * 32 + s * 16 + ln;
      ah[s] = load_frag_a(As_h + row * LDSA, hl, 0);
      al[s] = load_frag_a(As_l + row * LDSA, hl, 0);
    }
    v16bf bh[4], bl[4];
#pragma unroll
    for (int t = 0; t < 4; ++t) {
      int col = wn * 64 + t * 16 + ln;
      bh[t] = load_frag_b(Bs_h + col * LDSA, hl, 0);
      bl[t] = load_frag_b(Bs_l + col * LDSA, hl, 0);
    }
#pragma unroll
    for (int s = 0; s < 2; ++s)
#pragma unroll
      for (int t = 0; t < 4; ++t) {
        acc[s][t] = wmma_bf16(ah[s], bh[t], acc[s][t]);
        acc[s][t] = wmma_bf16(ah[s], bl[t], acc[s][t]);
        acc[s][t] = wmma_bf16(al[s], bh[t], acc[s][t]);
      }
  }

  // ---- branch-free epilogue (EPI is a template constant) ----
#pragma unroll
  for (int s = 0; s < 2; ++s) {
    int mg = bm0 + wm * 32 + s * 16 + hl * 8;
#pragma unroll
    for (int t = 0; t < 4; ++t) {
      int ng = bn0 + wn * 64 + t * 16 + ln;
      float bvv = bias[ng];
#pragma unroll
      for (int r = 0; r < 8; ++r) {
        float v = acc[s][t][r] + bvv;
        int row = mg + r;
        if (EPI == 1)      v *= scale;
        else if (EPI == 2) v = 0.5f * v * (1.0f + erff(v * 0.70710678118654752f));
        else if (EPI == 3) v += resid[(size_t)row * N + ng];
        C[(size_t)row * N + ng] = v;
      }
    }
  }
}

// ==========================================================================
// Flash attention: per (64-query tile, head). 128 threads = 4 waves,
// each wave owns 16 query rows. Streams 64-key tiles; K and V share LDS.
// ==========================================================================
#define APAD 72   // 64 data + 8 pad ushorts (144B, 16B aligned)

__global__ __launch_bounds__(128) void attn_flash(
    const float* __restrict__ Q, const float* __restrict__ K,
    const float* __restrict__ V, const float* __restrict__ mask,
    float* __restrict__ O, int Bsz, int Hn, int Ssz, int Dm)
{
  __shared__ u16 Qh[64 * APAD], Ql[64 * APAD];
  __shared__ u16 KVh[64 * APAD], KVl[64 * APAD];   // K tile, then reused for V
  __shared__ u16 Ph[64 * APAD], Pl[64 * APAD];     // wave-private 16-row blocks

  const int tid  = threadIdx.x;
  const int lane = tid & 31;
  const int wid  = tid >> 5;            // 0..3
  const int hl   = (lane >> 4) & 1;
  const int ln   = lane & 15;

  const int n  = blockIdx.y;            // head index in [0, B*H)
  const int b  = n / Hn;
  const int h  = n % Hn;
  const int RS = Bsz * Dm;              // row stride in (S,B,D) layout
  const int q0 = blockIdx.x * 64;

  const float* Qp = Q + b * Dm + h * 64;
  const float* Kp = K + b * Dm + h * 64;
  const float* Vp = V + b * Dm + h * 64;
  float*       Op = O + b * Dm + h * 64;

  // ---- load Q tile 64x64 ----
#pragma unroll
  for (int i = 0; i < 8; ++i) {
    int f  = tid + i * 128;             // 1024 float4s
    int r  = f >> 4;                    // 16 float4 per row
    int c4 = (f & 15) << 2;
    float4 v = *(const float4*)(Qp + (size_t)(q0 + r) * RS + c4);
    float a0[4] = {v.x, v.y, v.z, v.w};
#pragma unroll
    for (int j = 0; j < 4; ++j) {
      u16 hh = f2bf(a0[j]);
      Qh[r * APAD + c4 + j] = hh;
      Ql[r * APAD + c4 + j] = f2bf(a0[j] - bf2f(hh));
    }
  }
  __syncthreads();

  // preload Q fragments (rows wid*16 .. wid*16+15), 2 K-steps over HD=64
  v16bf aqh[2], aql[2];
  {
    const u16* qrh = Qh + (wid * 16 + ln) * APAD;
    const u16* qrl = Ql + (wid * 16 + ln) * APAD;
#pragma unroll
    for (int ks = 0; ks < 2; ++ks) {
      aqh[ks] = load_frag_a(qrh, hl, ks * 32);
      aql[ks] = load_frag_a(qrl, hl, ks * 32);
    }
  }

  float m_run[8], l_run[8];
  v8f acc_o[4];
#pragma unroll
  for (int r = 0; r < 8; ++r) { m_run[r] = -3.0e38f; l_run[r] = 0.0f; }
#pragma unroll
  for (int t = 0; t < 4; ++t) acc_o[t] = (v8f){};

  const int nKT = Ssz / 64;
  for (int j = 0; j < nKT; ++j) {
    const int k0 = j * 64;
    __syncthreads();
    // ---- load K tile 64x64 as [key][hd] ----
#pragma unroll
    for (int i = 0; i < 8; ++i) {
      int f  = tid + i * 128;
      int r  = f >> 4;
      int c4 = (f & 15) << 2;
      float4 v = *(const float4*)(Kp + (size_t)(k0 + r) * RS + c4);
      float a0[4] = {v.x, v.y, v.z, v.w};
#pragma unroll
      for (int jj = 0; jj < 4; ++jj) {
        u16 hh = f2bf(a0[jj]);
        KVh[r * APAD + c4 + jj] = hh;
        KVl[r * APAD + c4 + jj] = f2bf(a0[jj] - bf2f(hh));
      }
    }
    __syncthreads();

    // ---- scores S = Q * K^T (B operand = K rows as [n=key][k=hd]) ----
    v8f sacc[4];
#pragma unroll
    for (int t = 0; t < 4; ++t) sacc[t] = (v8f){};
#pragma unroll
    for (int t = 0; t < 4; ++t) {
      const u16* rph = KVh + (t * 16 + ln) * APAD;
      const u16* rpl = KVl + (t * 16 + ln) * APAD;
#pragma unroll
      for (int ks = 0; ks < 2; ++ks) {
        v16bf kbh = load_frag_b(rph, hl, ks * 32);
        v16bf kbl = load_frag_b(rpl, hl, ks * 32);
        sacc[t] = wmma_bf16(aqh[ks], kbh, sacc[t]);
        sacc[t] = wmma_bf16(aqh[ks], kbl, sacc[t]);
        sacc[t] = wmma_bf16(aql[ks], kbh, sacc[t]);
      }
    }

    // ---- padding mask (1 = pad -> -1e9 on that key column) ----
    float mk[4];
#pragma unroll
    for (int t = 0; t < 4; ++t) mk[t] = -1.0e9f * mask[b * Ssz + k0 + t * 16 + ln];

    // ---- online softmax (row = r + 8*hl within each wave's 16 rows) ----
#pragma unroll
    for (int r = 0; r < 8; ++r) {
      float mloc = sacc[0][r] + mk[0];
      mloc = fmaxf(mloc, sacc[1][r] + mk[1]);
      mloc = fmaxf(mloc, sacc[2][r] + mk[2]);
      mloc = fmaxf(mloc, sacc[3][r] + mk[3]);
#pragma unroll
      for (int d = 1; d < 16; d <<= 1) mloc = fmaxf(mloc, __shfl_xor(mloc, d, 32));
      float mnew  = fmaxf(m_run[r], mloc);
      float alpha = expf(m_run[r] - mnew);
      m_run[r] = mnew;
      float rowsum = 0.0f;
#pragma unroll
      for (int t = 0; t < 4; ++t) {
        float p = expf(sacc[t][r] + mk[t] - mnew);
        sacc[t][r] = p;
        rowsum += p;
      }
#pragma unroll
      for (int d = 1; d < 16; d <<= 1) rowsum += __shfl_xor(rowsum, d, 32);
      l_run[r] = l_run[r] * alpha + rowsum;
#pragma unroll
      for (int t = 0; t < 4; ++t) acc_o[t][r] *= alpha;
    }

    // ---- stage P (C layout -> LDS rows) ----
#pragma unroll
    for (int t = 0; t < 4; ++t)
#pragma unroll
      for (int r = 0; r < 8; ++r) {
        int m = wid * 16 + r + hl * 8;
        float p = sacc[t][r];
        u16 hh = f2bf(p);
        Ph[m * APAD + t * 16 + ln] = hh;
        Pl[m * APAD + t * 16 + ln] = f2bf(p - bf2f(hh));
      }
    __syncthreads();

    // ---- load V tile transposed [hd][key] into KV buffers ----
#pragma unroll
    for (int i = 0; i < 8; ++i) {
      int f  = tid + i * 128;
      int r  = f >> 4;                  // key index
      int c4 = (f & 15) << 2;           // hd
      float4 v = *(const float4*)(Vp + (size_t)(k0 + r) * RS + c4);
      float a0[4] = {v.x, v.y, v.z, v.w};
#pragma unroll
      for (int jj = 0; jj < 4; ++jj) {
        u16 hh = f2bf(a0[jj]);
        KVh[(c4 + jj) * APAD + r] = hh;
        KVl[(c4 + jj) * APAD + r] = f2bf(a0[jj] - bf2f(hh));
      }
    }
    __syncthreads();

    // ---- O += P * V ----
#pragma unroll
    for (int ks = 0; ks < 2; ++ks) {
      v16bf pah = load_frag_a(Ph + (wid * 16 + ln) * APAD, hl, ks * 32);
      v16bf pal = load_frag_a(Pl + (wid * 16 + ln) * APAD, hl, ks * 32);
#pragma unroll
      for (int t = 0; t < 4; ++t) {
        v16bf vbh = load_frag_b(KVh + (t * 16 + ln) * APAD, hl, ks * 32);
        v16bf vbl = load_frag_b(KVl + (t * 16 + ln) * APAD, hl, ks * 32);
        acc_o[t] = wmma_bf16(pah, vbh, acc_o[t]);
        acc_o[t] = wmma_bf16(pah, vbl, acc_o[t]);
        acc_o[t] = wmma_bf16(pal, vbh, acc_o[t]);
      }
    }
  }

  // ---- write O / l ----
#pragma unroll
  for (int t = 0; t < 4; ++t) {
    int hd = t * 16 + ln;
#pragma unroll
    for (int r = 0; r < 8; ++r) {
      int qrow = q0 + wid * 16 + r + hl * 8;
      Op[(size_t)qrow * RS + hd] = acc_o[t][r] / l_run[r];
    }
  }
}

// ==========================================================================
// Row LayerNorm over D=1024, 256 threads (4 floats/thread).
// ==========================================================================
__global__ __launch_bounds__(256) void layernorm_k(
    const float* __restrict__ X, const float* __restrict__ g,
    const float* __restrict__ bt, float* __restrict__ Y)
{
  const int Dm = 1024;
  const int row = blockIdx.x;
  const int lane = threadIdx.x & 31, wid = threadIdx.x >> 5;
  __shared__ float red[8];

  float4 v = ((const float4*)(X + (size_t)row * Dm))[threadIdx.x];
  float s = v.x + v.y + v.z + v.w;
#pragma unroll
  for (int d = 16; d >= 1; d >>= 1) s += __shfl_xor(s, d, 32);
  if (lane == 0) red[wid] = s;
  __syncthreads();
  float tot = 0.0f;
#pragma unroll
  for (int i = 0; i < 8; ++i) tot += red[i];
  float mean = tot * (1.0f / 1024.0f);
  __syncthreads();

  float dx0 = v.x - mean, dx1 = v.y - mean, dx2 = v.z - mean, dx3 = v.w - mean;
  float sq = dx0 * dx0 + dx1 * dx1 + dx2 * dx2 + dx3 * dx3;
#pragma unroll
  for (int d = 16; d >= 1; d >>= 1) sq += __shfl_xor(sq, d, 32);
  if (lane == 0) red[wid] = sq;
  __syncthreads();
  float tot2 = 0.0f;
#pragma unroll
  for (int i = 0; i < 8; ++i) tot2 += red[i];
  float rstd = rsqrtf(tot2 * (1.0f / 1024.0f) + 1e-5f);

  float4 gg = ((const float4*)g)[threadIdx.x];
  float4 bb = ((const float4*)bt)[threadIdx.x];
  float4 y;
  y.x = dx0 * rstd * gg.x + bb.x;
  y.y = dx1 * rstd * gg.y + bb.y;
  y.z = dx2 * rstd * gg.z + bb.z;
  y.w = dx3 * rstd * gg.w + bb.w;
  ((float4*)(Y + (size_t)row * Dm))[threadIdx.x] = y;
}

// ==========================================================================
extern "C" void kernel_launch(void* const* d_in, const int* in_sizes, int n_in,
                              void* d_out, int out_size, void* d_ws, size_t ws_size,
                              hipStream_t stream) {
  (void)in_sizes; (void)n_in; (void)out_size; (void)ws_size;
  const int S = 2048, Bz = 2, D = 1024, H = 16, FFN = 4096;
  const int M = S * Bz;                     // 4096 token rows
  const float scaling = 0.125f;             // (D/H)^-0.5 = 64^-0.5

  const float* x    = (const float*)d_in[0];
  const float* mask = (const float*)d_in[1];
  const float* Wq = (const float*)d_in[2];  const float* bq = (const float*)d_in[3];
  const float* Wk = (const float*)d_in[4];  const float* bk = (const float*)d_in[5];
  const float* Wv = (const float*)d_in[6];  const float* bv = (const float*)d_in[7];
  const float* Wo = (const float*)d_in[8];  const float* bo = (const float*)d_in[9];
  const float* g1 = (const float*)d_in[10]; const float* be1 = (const float*)d_in[11];
  const float* W1 = (const float*)d_in[12]; const float* b1 = (const float*)d_in[13];
  const float* W2 = (const float*)d_in[14]; const float* b2 = (const float*)d_in[15];
  const float* g2 = (const float*)d_in[16]; const float* be2 = (const float*)d_in[17];
  float* out = (float*)d_out;

  char* ws = (char*)d_ws;
  const size_t MB16 = (size_t)1 << 24;      // 16 MB
  float* xq = (float*)(ws + 0 * MB16);      // (M, D)
  float* xk = (float*)(ws + 1 * MB16);
  float* xv = (float*)(ws + 2 * MB16);
  float* ao = (float*)(ws + 3 * MB16);      // attention output (M, D)
  float* hh = (float*)(ws + 4 * MB16);      // FFN hidden (M, FFN) = 64 MB
  float* tmp1 = xq;                         // pre-LN1 (q no longer needed)
  float* x1   = xk;                         // LN1 output
  float* tmp2 = xv;                         // pre-LN2

  dim3 blk(256);

  // QKV projections
  gemm_bf16x3<1><<<dim3(D / BN, M / BM), blk, 0, stream>>>(x, Wq, bq, nullptr, xq, M, D, D, scaling);
  gemm_bf16x3<0><<<dim3(D / BN, M / BM), blk, 0, stream>>>(x, Wk, bk, nullptr, xk, M, D, D, 1.0f);
  gemm_bf16x3<0><<<dim3(D / BN, M / BM), blk, 0, stream>>>(x, Wv, bv, nullptr, xv, M, D, D, 1.0f);

  // fused attention
  attn_flash<<<dim3(S / 64, Bz * H), dim3(128), 0, stream>>>(xq, xk, xv, mask, ao, Bz, H, S, D);

  // output projection + residual, LN1
  gemm_bf16x3<3><<<dim3(D / BN, M / BM), blk, 0, stream>>>(ao, Wo, bo, x, tmp1, M, D, D, 1.0f);
  layernorm_k<<<dim3(M), blk, 0, stream>>>(tmp1, g1, be1, x1);

  // FFN: gelu(x1@W1+b1) @ W2 + b2 + x1, LN2 -> out
  gemm_bf16x3<2><<<dim3(FFN / BN, M / BM), blk, 0, stream>>>(x1, W1, b1, nullptr, hh, M, FFN, D, 1.0f);
  gemm_bf16x3<3><<<dim3(D / BN, M / BM), blk, 0, stream>>>(hh, W2, b2, x1, tmp2, M, D, FFN, 1.0f);
  layernorm_k<<<dim3(M), blk, 0, stream>>>(tmp2, g2, be2, out);
}
